// GRU_CNN_Attention_54649163875109
// MI455X (gfx1250) — matrix operand order repaired and checked by reading, not compiled, not run on hardware
//
#include <hip/hip_runtime.h>
#include <hip/hip_bf16.h>
#include <stdint.h>

// ---------------------------------------------------------------------------
// Model dims
// ---------------------------------------------------------------------------
#define BB    16
#define SS    64
#define WW    64
#define NROW  1024        // BB*SS
#define TTOK  65536       // NROW*WW
#define EMBD  300
#define KEPAD 320         // EMB padded to multiple of 32
#define HH    256
#define H2    512
#define H3    768

typedef __attribute__((ext_vector_type(16))) _Float16 v16h;
typedef __attribute__((ext_vector_type(8)))  _Float16 v8h;
typedef __attribute__((ext_vector_type(8)))  float    v8f;

__device__ __forceinline__ float sigf(float x) { return 1.0f / (1.0f + expf(-x)); }

// ---------------------------------------------------------------------------
// WMMA GEMM:  C[M,N](f32 or f16) = A[M,K]f16 * W[N,K]f16^T + bias[N]
//   A row-major (lda = K), W row-major weight layout (ldb = K) -- the
//   untransposed PyTorch weight layout is exactly the WMMA B-fragment layout
//   (lane half h: K 8h..8h+7 and 16+8h..16+8h+7).  K must be a multiple of 32.
// flags: bit0 -> store f16, bit1 -> tanh epilogue
// Block: 256 threads = 8 waves; block tile 64(M) x 128(N);
// wave tile 16x64: 4 accumulators, A fragment reused 4x; all B fragments
// fetched before the WMMA chain so the 4 matrix ops issue back-to-back.
// ---------------------------------------------------------------------------
__global__ __launch_bounds__(256) void k_wmma_gemm(
    const _Float16* __restrict__ A, int lda,
    const _Float16* __restrict__ Bm, int ldb,
    const float* __restrict__ bias,
    void* __restrict__ C, int ldc,
    int M, int N, int K, int flags)
{
  __shared__ __align__(16) _Float16 Al[64 * 40];    // A tile 64 rows x 32 K
  __shared__ __align__(16) _Float16 Bl[128 * 40];   // W tile 128 n-rows x 32 K

  const int tid  = threadIdx.x;
  const int lane = tid & 31;
  const int wave = tid >> 5;
  const int bm = blockIdx.y * 64;
  const int bn = blockIdx.x * 128;

  const int mt  = wave >> 1;            // 0..3 : 16-row M strip
  const int ntb = (wave & 1) * 4;       // base 16-col tile within 128 (0 or 4)

  v8f acc[4] = {{}, {}, {}, {}};

  for (int k0 = 0; k0 < K; k0 += 32) {
    __syncthreads();
    // ---- stage A tile: 64 rows x 32 halves (b128 stores)
    {
      int r = tid >> 2;
      int c = (tid & 3) * 8;
      int gm = bm + r;
      v8h av = {};
      if (gm < M) av = *(const v8h*)(A + (size_t)gm * lda + k0 + c);
      *(v8h*)&Al[r * 40 + c] = av;
    }
    // ---- stage W tile: 128 n-rows x 32 halves (b128 stores, no transpose)
#pragma unroll
    for (int i = 0; i < 2; ++i) {
      int idx = tid + i * 256;
      int r = idx >> 2;                 // 0..127
      int c = (idx & 3) * 8;
      int gn = bn + r;
      v8h bv = {};
      if (gn < N) bv = *(const v8h*)(Bm + (size_t)gn * ldb + k0 + c);
      *(v8h*)&Bl[r * 40 + c] = bv;
    }
    // ---- prefetch next K slab into near caches (global_prefetch_b8)
    if (k0 + 32 < K) {
      int gm = bm + (tid >> 2); if (gm >= M) gm = M - 1;
      __builtin_prefetch((const void*)(A + (size_t)gm * lda + (k0 + 32) + (tid & 3) * 8), 0, 3);
      int gn = bn + (tid >> 1); if (gn >= N) gn = N - 1;
      __builtin_prefetch((const void*)(Bm + (size_t)gn * ldb + (k0 + 32) + (tid & 1) * 16), 0, 3);
    }
    __syncthreads();

    // ---- A fragment (16x32): lanes 0-15 K={0..7,16..23}, lanes 16-31 K={8..15,24..31}
    int mrow = mt * 16 + (lane & 15);
    int akb  = (lane >> 4) * 8;
    v8h alo = *(const v8h*)&Al[mrow * 40 + akb];
    v8h ahi = *(const v8h*)&Al[mrow * 40 + akb + 16];
    v16h afrag = __builtin_shufflevector(alo, ahi, 0,1,2,3,4,5,6,7,8,9,10,11,12,13,14,15);

    // ---- fetch all 4 B fragments first (one dscnt wait covers them all)
    int bkb = (lane >> 4) * 8;
    v16h bfrag[4];
#pragma unroll
    for (int j = 0; j < 4; ++j) {
      int nr = (ntb + j) * 16 + (lane & 15);
      v8h blo = *(const v8h*)&Bl[nr * 40 + bkb];
      v8h bhi = *(const v8h*)&Bl[nr * 40 + bkb + 16];
      bfrag[j] = __builtin_shufflevector(blo, bhi, 0,1,2,3,4,5,6,7,8,9,10,11,12,13,14,15);
    }
    // ---- issue the 4 WMMAs back-to-back
#pragma unroll
    for (int j = 0; j < 4; ++j)
      acc[j] = __builtin_amdgcn_wmma_f32_16x16x32_f16(false, afrag, false, bfrag[j],
                                                      (short)0, acc[j], false, false);
  }

  // ---- epilogue: lane holds rows m = mbase+v (v=0..7), col n = lane&15
  int mbase = bm + mt * 16 + (lane >> 4) * 8;
#pragma unroll
  for (int j = 0; j < 4; ++j) {
    int n = bn + (ntb + j) * 16 + (lane & 15);
    if (n >= N) continue;
    float bi = bias ? bias[n] : 0.f;
#pragma unroll
    for (int v = 0; v < 8; ++v) {
      int m = mbase + v;
      if (m >= M) continue;
      float x = acc[j][v] + bi;
      if (flags & 2) x = tanhf(x);
      if (flags & 1) ((_Float16*)C)[(size_t)m * ldc + n] = (_Float16)x;
      else           ((float*)C)   [(size_t)m * ldc + n] = x;
    }
  }
}

// ---------------------------------------------------------------------------
// Utility kernels
// ---------------------------------------------------------------------------
__global__ void k_zero(uint32_t* p, size_t n) {
  size_t i = (size_t)blockIdx.x * blockDim.x + threadIdx.x;
  size_t st = (size_t)gridDim.x * blockDim.x;
  for (; i < n; i += st) p[i] = 0u;
}

// f32 -> f16 weight pack, keep [Nout][K] row-major, zero-pad K..Kpad
__global__ void k_pack_w16(_Float16* dst, const float* src, int Nout, int K, int Kpad) {
  int idx = blockIdx.x * 256 + threadIdx.x;
  if (idx >= Nout * Kpad) return;
  int n = idx / Kpad, k = idx - n * Kpad;
  float v = (k < K) ? src[(size_t)n * K + k] : 0.f;
  dst[(size_t)n * Kpad + k] = (_Float16)v;
}

__global__ void k_lens(const float* post_masks, const float* text_masks,
                       int* wlens, int* tlens) {
  int tid = blockIdx.x * 256 + threadIdx.x;
  if (tid < NROW) {
    float s = 0.f;
    for (int w = 0; w < WW; ++w) s += post_masks[tid * WW + w];
    wlens[tid] = (int)(s + 0.5f);
  } else if (tid < NROW + BB) {
    int b = tid - NROW; float s = 0.f;
    for (int i = 0; i < SS; ++i) s += text_masks[b * SS + i];
    tlens[b] = (int)(s + 0.5f);
  }
}

// Gather embeddings (fwd + length-reversed), f32 -> f16, K padded w/ zeros.
__global__ void k_embed(const int* x, const float* emb, const int* wlens,
                        _Float16* X, _Float16* Xr) {
  int t = blockIdx.x;               // token row = n*WW + w
  int n = t >> 6, w = t & 63;
  int len = wlens[n];
  int ws = len - 1 - w; ws = ws < 0 ? 0 : (ws > WW - 1 ? WW - 1 : ws);
  const float* e0 = emb + (size_t)x[t] * EMBD;
  const float* e1 = emb + (size_t)x[(n << 6) + ws] * EMBD;
  for (int j = threadIdx.x; j < KEPAD; j += blockDim.x) {
    X [(size_t)t * KEPAD + j] = (_Float16)((j < EMBD) ? e0[j] : 0.f);
    Xr[(size_t)t * KEPAD + j] = (_Float16)((j < EMBD) ? e1[j] : 0.f);
  }
}

// Word-level GRU gate step (one block per sequence row, thread = hidden dim)
__global__ void k_gate_word(const _Float16* __restrict__ gi, const float* __restrict__ gh,
                            float* hstate, _Float16* h16,
                            float* wout, _Float16* wout16,
                            const int* lens, int t, int dirBack) {
  int n = blockIdx.x, j = threadIdx.x;
  size_t tok = (size_t)n * WW + t;
  float gir = (float)gi[tok * H3 + j];
  float giz = (float)gi[tok * H3 + HH + j];
  float gin = (float)gi[tok * H3 + 2 * HH + j];
  const float* ghr = gh + (size_t)n * H3;
  float r  = sigf(gir + ghr[j]);
  float z  = sigf(giz + ghr[HH + j]);
  float nn = tanhf(gin + r * ghr[2 * HH + j]);
  float h  = (1.f - z) * nn + z * hstate[n * HH + j];
  hstate[n * HH + j] = h;
  h16[n * HH + j] = (_Float16)h;
  int len = lens[n];
  if (t < len) {
    int w = dirBack ? (len - 1 - t) : t;
    size_t o = ((size_t)n * WW + w) * H2 + (dirBack ? HH : 0) + j;
    wout[o]   = h;
    wout16[o] = (_Float16)h;
  }
}

// Sentence-level GRU gate step
__global__ void k_gate_sent(const float* __restrict__ gi, const float* __restrict__ gh,
                            float* hstate, _Float16* h16,
                            float* sout, const int* tlens, int t, int dirBack) {
  int b = blockIdx.x, j = threadIdx.x;
  size_t tok = (size_t)b * SS + t;
  float gir = gi[tok * H3 + j];
  float giz = gi[tok * H3 + HH + j];
  float gin = gi[tok * H3 + 2 * HH + j];
  const float* ghr = gh + (size_t)b * H3;
  float r  = sigf(gir + ghr[j]);
  float z  = sigf(giz + ghr[HH + j]);
  float nn = tanhf(gin + r * ghr[2 * HH + j]);
  float h  = (1.f - z) * nn + z * hstate[b * HH + j];
  hstate[b * HH + j] = h;
  h16[b * HH + j] = (_Float16)h;
  int len = tlens[b];
  if (t < len) {
    int s = dirBack ? (len - 1 - t) : t;
    sout[((size_t)b * SS + s) * H2 + (dirBack ? HH : 0) + j] = h;
  }
}

// scores[row] = att16[row,:] . q  (wave per row, shuffle reduce)
__global__ void k_scores(const _Float16* __restrict__ att, const float* __restrict__ q,
                         float* scores) {
  int gtid = blockIdx.x * blockDim.x + threadIdx.x;
  int row = gtid >> 5, lane = threadIdx.x & 31;
  if (row >= TTOK) return;
  const _Float16* a = att + (size_t)row * H2;
  float s = 0.f;
  for (int j = lane; j < H2; j += 32) s += (float)a[j] * q[j];
  for (int off = 16; off; off >>= 1) s += __shfl_xor(s, off, 32);
  if (lane == 0) scores[row] = s;
}

// Word attention softmax + masked renorm + weighted pool -> pair_vector
__global__ void k_word_attn(const float* __restrict__ scores,
                            const float* __restrict__ post_masks,
                            const float* __restrict__ wout, float* pair) {
  __shared__ float aw[WW];
  int n = blockIdx.x, tid = threadIdx.x;
  if (tid < WW) aw[tid] = scores[n * WW + tid];
  __syncthreads();
  if (tid == 0) {
    float mx = aw[0];
    for (int w = 1; w < WW; ++w) mx = fmaxf(mx, aw[w]);
    float s = 0.f;
    for (int w = 0; w < WW; ++w) { aw[w] = expf(aw[w] - mx); s += aw[w]; }
    float s2 = 0.f;
    for (int w = 0; w < WW; ++w) { aw[w] = (aw[w] / s) * post_masks[n * WW + w]; s2 += aw[w]; }
    float inv = 1.f / (s2 + 1e-4f);
    for (int w = 0; w < WW; ++w) aw[w] *= inv;
  }
  __syncthreads();
  for (int d = tid; d < H2; d += blockDim.x) {
    float acc = 0.f;
    for (int w = 0; w < WW; ++w) acc += aw[w] * wout[((size_t)n * WW + w) * H2 + d];
    pair[(size_t)n * H2 + d] = acc;
  }
}

// pair f32 -> f16 and length-reversed f16
__global__ void k_pack_pair(const float* pair, const int* tlens,
                            _Float16* p16, _Float16* pr16) {
  int t = blockIdx.x;               // b*SS + s
  int b = t >> 6, s = t & 63;
  int tl = tlens[b];
  int sr = tl - 1 - s; sr = sr < 0 ? 0 : (sr > SS - 1 ? SS - 1 : sr);
  for (int j = threadIdx.x; j < H2; j += blockDim.x) {
    p16 [(size_t)t * H2 + j] = (_Float16)pair[(size_t)t * H2 + j];
    pr16[(size_t)t * H2 + j] = (_Float16)pair[((size_t)(b * SS + sr)) * H2 + j];
  }
}

// Sentence attention pooling -> doc
__global__ void k_sent_attn(const float* __restrict__ sout, const float* __restrict__ attw,
                            const float* __restrict__ tmask, float* doc) {
  __shared__ float sc[SS];
  int b = blockIdx.x, tid = threadIdx.x;
  if (tid < SS) sc[tid] = 0.f;
  __syncthreads();
  int s = tid & 63, part = tid >> 6;       // 4 partial sums per row
  float p = 0.f;
  for (int j = part * 128; j < part * 128 + 128; ++j)
    p += sout[((size_t)b * SS + s) * H2 + j] * attw[j];
  atomicAdd(&sc[s], p);
  __syncthreads();
  if (tid == 0) {
    float mx = 0.f;
    for (int w = 0; w < SS; ++w) { sc[w] = fmaxf(sc[w], 0.f); mx = fmaxf(mx, sc[w]); }
    float ssum = 0.f;
    for (int w = 0; w < SS; ++w) { sc[w] = expf(sc[w] - mx); ssum += sc[w]; }
    float s2 = 0.f;
    for (int w = 0; w < SS; ++w) { sc[w] = sc[w] / ssum * tmask[b * SS + w]; s2 += sc[w]; }
    float inv = 1.f / s2;
    for (int w = 0; w < SS; ++w) sc[w] *= inv;
  }
  __syncthreads();
  for (int d = tid; d < H2; d += blockDim.x) {
    float acc = 0.f;
    for (int w = 0; w < SS; ++w) acc += sc[w] * sout[((size_t)b * SS + w) * H2 + d];
    doc[b * H2 + d] = acc;
  }
}

// Commonality over last <=35 valid posts
__global__ void k_comm(const float* __restrict__ pair, const float* __restrict__ tmask,
                       float* comm) {
  __shared__ float rec[SS];
  __shared__ float linv;
  int b = blockIdx.x, tid = threadIdx.x;
  if (tid == 0) {
    float tmp[SS]; float rc = 0.f, l = 0.f;
    for (int s = SS - 1; s >= 0; --s) { rc += tmask[b * SS + s]; tmp[s] = rc; }
    for (int s = 0; s < SS; ++s) {
      float r = tmask[b * SS + s] * ((tmp[s] <= 35.f) ? 1.f : 0.f);
      rec[s] = r; l += r;
    }
    linv = 1.f / (2.f * l);
  }
  __syncthreads();
  for (int d = tid; d < H2; d += blockDim.x) {
    float sv = 0.f, ss = 0.f;
    for (int s = 0; s < SS; ++s) {
      float r = rec[s], v = pair[((size_t)b * SS + s) * H2 + d];
      sv += r * v; ss += r * v * v;
    }
    float sp = (sv * sv - ss) * linv;
    comm[b * H2 + d] = sp > 0.f ? sp : 0.01f * sp;
  }
}

// Adaptive fusion + feature concat
__global__ void k_fusion(const float* doc, const float* comm,
                         const float* Wf, const float* bf,
                         const float* Wt, const float* bt,
                         const float* sout, float* feats) {
  int b = blockIdx.x, j = threadIdx.x;   // 512 threads
  float a1 = bf[j], a2 = bt[j];
  const float* dr = doc + b * H2;
  const float* cr = comm + b * H2;
  for (int k = 0; k < H2; ++k) {
    a1 += dr[k] * Wf[(size_t)j * H2 + k];
    a2 += cr[k] * Wt[(size_t)j * H2 + k];
  }
  float lam = sigf(a1 + a2);
  feats[(size_t)b * 1024 + j]       = (1.f - lam) * dr[j] + lam * cr[j];
  feats[(size_t)b * 1024 + H2 + j]  = sout[((size_t)b * SS + (SS - 1)) * H2 + j];
}

// FC + train-mode BatchNorm (batch of 16) + ReLU; one thread per out column
__global__ void k_mlp_bn(const float* in, int K, const float* Wm, const float* bias,
                         const float* g, const float* be, float* out, int J) {
  int j = blockIdx.x * blockDim.x + threadIdx.x;
  if (j >= J) return;
  float v[BB]; float mu = 0.f;
  const float* wr = Wm + (size_t)j * K;
  for (int b = 0; b < BB; ++b) {
    float a = bias[j];
    const float* row = in + (size_t)b * K;
    for (int k = 0; k < K; ++k) a += row[k] * wr[k];
    v[b] = a; mu += a;
  }
  mu *= (1.f / BB);
  float var = 0.f;
  for (int b = 0; b < BB; ++b) { float d = v[b] - mu; var += d * d; }
  var *= (1.f / BB);
  float inv = rsqrtf(var + 1e-5f);
  for (int b = 0; b < BB; ++b)
    out[(size_t)b * J + j] = fmaxf(g[j] * (v[b] - mu) * inv + be[j], 0.f);
}

__global__ void k_final(const float* h2, const float* W3, const float* b3, float* out) {
  int tid = threadIdx.x;
  if (tid >= 32) return;
  int b = tid >> 1, c = tid & 1;
  float a = b3[c];
  for (int k = 0; k < HH; ++k) a += h2[b * HH + k] * W3[c * HH + k];
  out[b * 2 + c] = a;
}

// ---------------------------------------------------------------------------
// Host orchestration
// ---------------------------------------------------------------------------
static void gemm(const _Float16* A, int lda, const _Float16* Bm, int ldb,
                 const float* bias, void* C, int ldc,
                 int M, int N, int K, int flags, hipStream_t s) {
  dim3 g((N + 127) / 128, (M + 63) / 64);
  k_wmma_gemm<<<g, 256, 0, s>>>(A, lda, Bm, ldb, bias, C, ldc, M, N, K, flags);
}

static void zero(void* p, size_t bytes, hipStream_t s) {
  size_t n = bytes / 4;
  int grid = (int)((n + 255) / 256); if (grid > 8192) grid = 8192;
  k_zero<<<grid, 256, 0, s>>>((uint32_t*)p, n);
}

extern "C" void kernel_launch(void* const* d_in, const int* in_sizes, int n_in,
                              void* d_out, int out_size, void* d_ws, size_t ws_size,
                              hipStream_t stream) {
  (void)in_sizes; (void)n_in; (void)out_size; (void)ws_size;

  // ---- inputs (setup_inputs dict order; params flattened in insertion order)
  const int*   x        = (const int*)  d_in[0];
  const float* tmask    = (const float*)d_in[1];
  const float* pmask    = (const float*)d_in[2];
  const float* emb      = (const float*)d_in[3];
  const float* wWih_f   = (const float*)d_in[4];
  const float* wWhh_f   = (const float*)d_in[5];
  const float* wbih_f   = (const float*)d_in[6];
  const float* wbhh_f   = (const float*)d_in[7];
  const float* wWih_b   = (const float*)d_in[8];
  const float* wWhh_b   = (const float*)d_in[9];
  const float* wbih_b   = (const float*)d_in[10];
  const float* wbhh_b   = (const float*)d_in[11];
  const float* sWih_f   = (const float*)d_in[12];
  const float* sWhh_f   = (const float*)d_in[13];
  const float* sbih_f   = (const float*)d_in[14];
  const float* sbhh_f   = (const float*)d_in[15];
  const float* sWih_b   = (const float*)d_in[16];
  const float* sWhh_b   = (const float*)d_in[17];
  const float* sbih_b   = (const float*)d_in[18];
  const float* sbhh_b   = (const float*)d_in[19];
  const float* fcW      = (const float*)d_in[20];
  const float* fcB      = (const float*)d_in[21];
  const float* wquery   = (const float*)d_in[22];
  const float* attw     = (const float*)d_in[23];
  const float* Wf       = (const float*)d_in[24];
  const float* bf       = (const float*)d_in[25];
  const float* Wt       = (const float*)d_in[26];
  const float* bt       = (const float*)d_in[27];
  const float* m1W      = (const float*)d_in[28];
  const float* m1b      = (const float*)d_in[29];
  const float* bn1g     = (const float*)d_in[30];
  const float* bn1b     = (const float*)d_in[31];
  const float* m2W      = (const float*)d_in[32];
  const float* m2b      = (const float*)d_in[33];
  const float* bn2g     = (const float*)d_in[34];
  const float* bn2b     = (const float*)d_in[35];
  const float* m3W      = (const float*)d_in[36];
  const float* m3b      = (const float*)d_in[37];
  float* out = (float*)d_out;

  // ---- workspace carve-up
  char* ws = (char*)d_ws; size_t off = 0;
  auto alloc = [&](size_t bytes) -> void* {
    void* p = ws + off; off += (bytes + 255) & ~(size_t)255; return p;
  };
  _Float16* pBih_f = (_Float16*)alloc((size_t)H3 * KEPAD * 2);
  _Float16* pBih_b = (_Float16*)alloc((size_t)H3 * KEPAD * 2);
  _Float16* pBhh_f = (_Float16*)alloc((size_t)H3 * HH * 2);
  _Float16* pBhh_b = (_Float16*)alloc((size_t)H3 * HH * 2);
  _Float16* pFC    = (_Float16*)alloc((size_t)H2 * H2 * 2);
  _Float16* psBih_f= (_Float16*)alloc((size_t)H3 * H2 * 2);
  _Float16* psBih_b= (_Float16*)alloc((size_t)H3 * H2 * 2);
  _Float16* psBhh_f= (_Float16*)alloc((size_t)H3 * HH * 2);
  _Float16* psBhh_b= (_Float16*)alloc((size_t)H3 * HH * 2);
  _Float16* X16    = (_Float16*)alloc((size_t)TTOK * KEPAD * 2);
  _Float16* Xr16   = (_Float16*)alloc((size_t)TTOK * KEPAD * 2);
  _Float16* gi_f   = (_Float16*)alloc((size_t)TTOK * H3 * 2);
  _Float16* gi_b   = (_Float16*)alloc((size_t)TTOK * H3 * 2);
  float*    gh_f   = (float*)   alloc((size_t)NROW * H3 * 4);
  float*    gh_b   = (float*)   alloc((size_t)NROW * H3 * 4);
  float*    hs_f   = (float*)   alloc((size_t)NROW * HH * 4);
  float*    hs_b   = (float*)   alloc((size_t)NROW * HH * 4);
  _Float16* h16_f  = (_Float16*)alloc((size_t)NROW * HH * 2);
  _Float16* h16_b  = (_Float16*)alloc((size_t)NROW * HH * 2);
  float*    wout   = (float*)   alloc((size_t)TTOK * H2 * 4);
  _Float16* wout16 = (_Float16*)alloc((size_t)TTOK * H2 * 2);
  _Float16* att16  = (_Float16*)alloc((size_t)TTOK * H2 * 2);
  float*    scores = (float*)   alloc((size_t)TTOK * 4);
  float*    pair   = (float*)   alloc((size_t)NROW * H2 * 4);
  _Float16* pair16 = (_Float16*)alloc((size_t)NROW * H2 * 2);
  _Float16* pairr16= (_Float16*)alloc((size_t)NROW * H2 * 2);
  float*    sgi_f  = (float*)   alloc((size_t)NROW * H3 * 4);
  float*    sgi_b  = (float*)   alloc((size_t)NROW * H3 * 4);
  float*    sgh_f  = (float*)   alloc((size_t)BB * H3 * 4);
  float*    sgh_b  = (float*)   alloc((size_t)BB * H3 * 4);
  float*    shs_f  = (float*)   alloc((size_t)BB * HH * 4);
  float*    shs_b  = (float*)   alloc((size_t)BB * HH * 4);
  _Float16* sh16_f = (_Float16*)alloc((size_t)BB * HH * 2);
  _Float16* sh16_b = (_Float16*)alloc((size_t)BB * HH * 2);
  float*    sout   = (float*)   alloc((size_t)BB * SS * H2 * 4);
  float*    doc    = (float*)   alloc((size_t)BB * H2 * 4);
  float*    comm   = (float*)   alloc((size_t)BB * H2 * 4);
  float*    feats  = (float*)   alloc((size_t)BB * 1024 * 4);
  float*    h1     = (float*)   alloc((size_t)BB * H2 * 4);
  float*    h2b    = (float*)   alloc((size_t)BB * HH * 4);
  int*      wlens  = (int*)     alloc((size_t)NROW * 4);
  int*      tlens  = (int*)     alloc((size_t)BB * 4);

  // ---- zero-init (masked-output buffers + GRU states)
  zero(wout,   (size_t)TTOK * H2 * 4, stream);
  zero(wout16, (size_t)TTOK * H2 * 2, stream);
  zero(sout,   (size_t)BB * SS * H2 * 4, stream);
  zero(hs_f,   (size_t)NROW * HH * 4, stream);
  zero(hs_b,   (size_t)NROW * HH * 4, stream);
  zero(h16_f,  (size_t)NROW * HH * 2, stream);
  zero(h16_b,  (size_t)NROW * HH * 2, stream);
  zero(shs_f,  (size_t)BB * HH * 4, stream);
  zero(shs_b,  (size_t)BB * HH * 4, stream);
  zero(sh16_f, (size_t)BB * HH * 2, stream);
  zero(sh16_b, (size_t)BB * HH * 2, stream);

  // ---- lengths + weight packing + embedding gather
  k_lens<<<(NROW + BB + 255) / 256, 256, 0, stream>>>(pmask, tmask, wlens, tlens);
  auto pack = [&](_Float16* dst, const float* src, int Nout, int K, int Kpad) {
    k_pack_w16<<<(Nout * Kpad + 255) / 256, 256, 0, stream>>>(dst, src, Nout, K, Kpad);
  };
  pack(pBih_f, wWih_f, H3, EMBD, KEPAD);
  pack(pBih_b, wWih_b, H3, EMBD, KEPAD);
  pack(pBhh_f, wWhh_f, H3, HH, HH);
  pack(pBhh_b, wWhh_b, H3, HH, HH);
  pack(pFC,    fcW,    H2, H2, H2);
  pack(psBih_f, sWih_f, H3, H2, H2);
  pack(psBih_b, sWih_b, H3, H2, H2);
  pack(psBhh_f, sWhh_f, H3, HH, HH);
  pack(psBhh_b, sWhh_b, H3, HH, HH);
  k_embed<<<TTOK, 256, 0, stream>>>(x, emb, wlens, X16, Xr16);

  // ---- word-level input projections (big WMMA GEMMs, f16 out)
  gemm(X16,  KEPAD, pBih_f, KEPAD, wbih_f, gi_f, H3, TTOK, H3, KEPAD, 1, stream);
  gemm(Xr16, KEPAD, pBih_b, KEPAD, wbih_b, gi_b, H3, TTOK, H3, KEPAD, 1, stream);

  // ---- word-level recurrences (both directions per step)
  for (int t = 0; t < WW; ++t) {
    gemm(h16_f, HH, pBhh_f, HH, wbhh_f, gh_f, H3, NROW, H3, HH, 0, stream);
    k_gate_word<<<NROW, HH, 0, stream>>>(gi_f, gh_f, hs_f, h16_f, wout, wout16, wlens, t, 0);
    gemm(h16_b, HH, pBhh_b, HH, wbhh_b, gh_b, H3, NROW, H3, HH, 0, stream);
    k_gate_word<<<NROW, HH, 0, stream>>>(gi_b, gh_b, hs_b, h16_b, wout, wout16, wlens, t, 1);
  }

  // ---- word attention: tanh FC (WMMA, tanh epilogue) -> scores -> pool
  gemm(wout16, H2, pFC, H2, fcB, att16, H2, TTOK, H2, H2, 3, stream);
  k_scores<<<(TTOK * 32 + 255) / 256, 256, 0, stream>>>(att16, wquery, scores);
  k_word_attn<<<NROW, 256, 0, stream>>>(scores, pmask, wout, pair);
  k_pack_pair<<<NROW, 256, 0, stream>>>(pair, tlens, pair16, pairr16);

  // ---- sentence-level input projections
  gemm(pair16,  H2, psBih_f, H2, sbih_f, sgi_f, H3, NROW, H3, H2, 0, stream);
  gemm(pairr16, H2, psBih_b, H2, sbih_b, sgi_b, H3, NROW, H3, H2, 0, stream);

  // ---- sentence-level recurrences
  for (int t = 0; t < SS; ++t) {
    gemm(sh16_f, HH, psBhh_f, HH, sbhh_f, sgh_f, H3, BB, H3, HH, 0, stream);
    k_gate_sent<<<BB, HH, 0, stream>>>(sgi_f, sgh_f, shs_f, sh16_f, sout, tlens, t, 0);
    gemm(sh16_b, HH, psBhh_b, HH, sbhh_b, sgh_b, H3, BB, H3, HH, 0, stream);
    k_gate_sent<<<BB, HH, 0, stream>>>(sgi_b, sgh_b, shs_b, sh16_b, sout, tlens, t, 1);
  }

  // ---- pooling, commonality, fusion, MLP head
  k_sent_attn<<<BB, 256, 0, stream>>>(sout, attw, tmask, doc);
  k_comm<<<BB, 256, 0, stream>>>(pair, tmask, comm);
  k_fusion<<<BB, H2, 0, stream>>>(doc, comm, Wf, bf, Wt, bt, sout, feats);
  k_mlp_bn<<<2, 256, 0, stream>>>(feats, 1024, m1W, m1b, bn1g, bn1b, h1, H2);
  k_mlp_bn<<<1, 256, 0, stream>>>(h1, H2, m2W, m2b, bn2g, bn2b, h2b, HH);
  k_final<<<1, 32, 0, stream>>>(h2b, m3W, m3b, out);
}
